// MultiHeadAttention_12197707120982
// MI455X (gfx1250) — matrix operand order repaired and checked by reading, not compiled, not run on hardware
//
#include <hip/hip_runtime.h>
#include <math.h>

// B=4, T=2048, C=1024, H=16, D=64.  M = B*T = 8192.
typedef _Float16 v16h __attribute__((ext_vector_type(16)));
typedef float    v8f  __attribute__((ext_vector_type(8)));

#define WMMA_F32_F16(a, b, c) \
  __builtin_amdgcn_wmma_f32_16x16x32_f16(false, (a), false, (b), (short)0, (c), false, false)

// ---- CDNA5 async global->LDS copy (ASYNCcnt) -------------------------------
__device__ __forceinline__ unsigned lds_off_of(const void* p) {
  // generic LDS address: low 32 bits are the byte offset within LDS
  return (unsigned)(unsigned long long)p;
}
__device__ __forceinline__ void async_copy_b128(unsigned lds_byte_off,
                                                unsigned long long sbase,
                                                unsigned gbyte_off) {
  asm volatile("global_load_async_to_lds_b128 %0, %1, %2"
               :: "v"(lds_byte_off), "v"(gbyte_off), "s"(sbase)
               : "memory");
}
__device__ __forceinline__ void wait_async_all() {
  asm volatile("s_wait_asynccnt 0" ::: "memory");
}

// ---------------------------------------------------------------------------
// Pre-convert fp32 -> f16 (elementwise, vectorized)
// ---------------------------------------------------------------------------
__global__ __launch_bounds__(256) void cvt_f16_kernel(const float* __restrict__ in,
                                                      _Float16* __restrict__ outp) {
  int i = (blockIdx.x * 256 + threadIdx.x) * 4;
  float4 f = *(const float4*)(in + i);
  union { _Float16 h[4]; uint2 u; } pk;
  pk.h[0] = (_Float16)f.x; pk.h[1] = (_Float16)f.y;
  pk.h[2] = (_Float16)f.z; pk.h[3] = (_Float16)f.w;
  *(uint2*)(outp + i) = pk.u;
}

// ---------------------------------------------------------------------------
// Pre-convert + transpose fp32 [rows, cols] -> f16 [cols, rows]
// ---------------------------------------------------------------------------
__global__ __launch_bounds__(256) void cvt_transpose_kernel(const float* __restrict__ in,
                                                            _Float16* __restrict__ outp,
                                                            int rows, int cols) {
  __shared__ _Float16 tile[32][34];
  const int c0 = blockIdx.x * 32;
  const int r0 = blockIdx.y * 32;
  const int tx = threadIdx.x & 31;
  const int ty = (threadIdx.x >> 5) * 4;
#pragma unroll
  for (int e = 0; e < 4; ++e)
    tile[ty + e][tx] = (_Float16)in[(r0 + ty + e) * cols + c0 + tx];
  __syncthreads();
#pragma unroll
  for (int e = 0; e < 4; ++e)
    outp[(c0 + ty + e) * rows + r0 + tx] = tile[tx][ty + e];
}

// ---------------------------------------------------------------------------
// Kernel 1: qkv = x16 @ w_attn + b_attn.  A [8192,1024] f16, B^T [3072,1024]
// f16 (n-major).  Block tile 64x128, K staged 32 at a time, double-buffered
// async-to-LDS staging. 8 waves, wave tile 16x64 (4 WMMA accumulators).
// q,k stored [B,H,T,D] (q pre-scaled 1/8); v stored [B,H,D,T].
// ---------------------------------------------------------------------------
__global__ __launch_bounds__(256) void qkv_gemm_kernel(
    const _Float16* __restrict__ x,    // [8192,1024]
    const _Float16* __restrict__ wT,   // [3072,1024]
    const float* __restrict__ bias,    // [3072]
    _Float16* __restrict__ qh,         // [B,H,T,D]
    _Float16* __restrict__ kh,         // [B,H,T,D]
    _Float16* __restrict__ vh)         // [B,H,D,T]
{
  __shared__ __align__(16) _Float16 Alds[2][64 * 48];   // [m][k], 96B row stride
  __shared__ __align__(16) _Float16 Blds[2][128 * 48];  // [n][k]

  const int tid   = threadIdx.x;
  const int lane  = tid & 31;
  const int wave  = tid >> 5;
  const int m0    = blockIdx.y * 64;
  const int n0    = blockIdx.x * 128;
  const int mw    = (wave >> 1) * 16;
  const int nw    = (wave & 1) * 64;
  const int lhalf = lane >> 4;
  const int l15   = lane & 15;

  const unsigned long long xs = (unsigned long long)(size_t)x;
  const unsigned long long wb = (unsigned long long)(size_t)wT;
  const unsigned aoff0 = lds_off_of(&Alds[0][0]);
  const unsigned boff0 = lds_off_of(&Blds[0][0]);

  auto stage = [&](int k0, int b) {
    {  // A: 64 rows x 4 chunks(16B) = 256 slots, 1/thread
      int r = tid >> 2, cg = tid & 3;
      async_copy_b128(aoff0 + (unsigned)(b * 6144 + r * 96 + cg * 16), xs,
                      (unsigned)(((m0 + r) * 1024 + k0 + cg * 8) * 2));
    }
#pragma unroll
    for (int it = 0; it < 2; ++it) {  // B: 128 rows x 4 chunks = 512 slots
      int s = tid + it * 256;
      int n = s >> 2, cg = s & 3;
      async_copy_b128(boff0 + (unsigned)(b * 12288 + n * 96 + cg * 16), wb,
                      (unsigned)(((n0 + n) * 1024 + k0 + cg * 8) * 2));
    }
  };

  v8f acc[4];
#pragma unroll
  for (int t = 0; t < 4; ++t)
#pragma unroll
    for (int j = 0; j < 8; ++j) acc[t][j] = 0.0f;

  stage(0, 0);
  wait_async_all();
  __syncthreads();

  for (int kt = 0; kt < 32; ++kt) {
    const int cur = kt & 1;
    if (kt + 1 < 32) stage((kt + 1) * 32, cur ^ 1);  // overlap DMA with WMMA

    const _Float16* Ab = &Alds[cur][0];
    const _Float16* Bb = &Blds[cur][0];
    v16h af;
    {
      const _Float16* p = &Ab[(mw + l15) * 48 + lhalf * 8];
#pragma unroll
      for (int i = 0; i < 8; ++i) { af[i] = p[i]; af[8 + i] = p[16 + i]; }
    }
#pragma unroll
    for (int t = 0; t < 4; ++t) {
      v16h bf;
      const _Float16* p = &Bb[(nw + t * 16 + l15) * 48 + lhalf * 8];
#pragma unroll
      for (int i = 0; i < 8; ++i) { bf[i] = p[i]; bf[8 + i] = p[16 + i]; }
      acc[t] = WMMA_F32_F16(af, bf, acc[t]);
    }
    wait_async_all();
    __syncthreads();
  }

  // Epilogue: per-tile-uniform scatter target (branch-light)
  const int mrow0 = m0 + mw + lhalf * 8;   // 8 consecutive rows, same batch
  const int bb    = mrow0 >> 11;
  const int tt0   = mrow0 & 2047;
#pragma unroll
  for (int t = 0; t < 4; ++t) {
    const int nbase = n0 + nw + t * 16;    // uniform per tile
    const int sec   = nbase >> 10;         // 0=q 1=k 2=v
    const int cc    = nbase & 1023;
    const int h     = cc >> 6;
    const int dbase = cc & 63;
    const float bv  = bias[nbase + l15];
    if (sec == 2) {
      // V: [B,H,D,T] -> 8 row-consecutive halves per lane, one b128 store
      union { _Float16 hh[8]; uint4 u; } pk;
#pragma unroll
      for (int j = 0; j < 8; ++j) pk.hh[j] = (_Float16)(acc[t][j] + bv);
      _Float16* dst = vh + ((size_t)(bb * 16 + h) * 64 + dbase + l15) * 2048 + tt0;
      *(uint4*)dst = pk.u;
    } else {
      const float scl = (sec == 0) ? 0.125f : 1.0f;   // fold 1/sqrt(D) into q
      _Float16* dst = (sec == 0) ? qh : kh;
#pragma unroll
      for (int j = 0; j < 8; ++j) {
        size_t off = ((size_t)(bb * 16 + h) * 2048 + tt0 + j) * 64 + dbase + l15;
        dst[off] = (_Float16)((acc[t][j] + bv) * scl);
      }
    }
  }
}

// ---------------------------------------------------------------------------
// Kernel 2: causal flash attention. Grid (qb, b*h), 128 threads (4 waves),
// each wave owns a 16-row query tile. Key tiles of 32 double-buffered via
// async-to-LDS (K row-major, V pre-transposed in global so both are copies).
// ---------------------------------------------------------------------------
__global__ __launch_bounds__(128) void flash_attn_kernel(
    const _Float16* __restrict__ q,   // [B,H,T,D], pre-scaled
    const _Float16* __restrict__ k,   // [B,H,T,D]
    const _Float16* __restrict__ v,   // [B,H,D,T]
    _Float16* __restrict__ attn_out)  // [B,T,C]
{
  __shared__ __align__(16) _Float16 Klds[2][32 * 72];   // [key][d], 144B rows
  __shared__ __align__(16) _Float16 Vlds[2][64 * 48];   // [d][key], 96B rows
  __shared__ __align__(16) _Float16 Plds[4][16 * 48];   // per-wave P tile

  const int tid   = threadIdx.x;
  const int lane  = tid & 31;
  const int wave  = tid >> 5;
  const int lhalf = lane >> 4;
  const int l15   = lane & 15;
  const int bh    = blockIdx.y;            // [0,64)
  const int qb    = blockIdx.x;            // [0,32)
  const int qt    = qb * 64 + wave * 16;

  const _Float16* qbase = q + (size_t)bh * 2048 * 64;
  const unsigned long long kbp = (unsigned long long)(size_t)(k + (size_t)bh * 2048 * 64);
  const unsigned long long vbp = (unsigned long long)(size_t)(v + (size_t)bh * 64 * 2048);
  const unsigned koff0 = lds_off_of(&Klds[0][0]);
  const unsigned voff0 = lds_off_of(&Vlds[0][0]);

  auto stage = [&](int kb, int b) {
#pragma unroll
    for (int it = 0; it < 2; ++it) {  // K: 32 keys x 8 chunks = 256 slots
      int s = tid + it * 128;
      int key = s >> 3, ch = s & 7;
      async_copy_b128(koff0 + (unsigned)(b * 4608 + key * 144 + ch * 16), kbp,
                      (unsigned)(((kb + key) * 64 + ch * 8) * 2));
    }
#pragma unroll
    for (int it = 0; it < 2; ++it) {  // V: 64 d x 4 chunks = 256 slots
      int s = tid + it * 128;
      int d = s >> 2, ch = s & 3;
      async_copy_b128(voff0 + (unsigned)(b * 6144 + d * 96 + ch * 16), vbp,
                      (unsigned)((d * 2048 + kb + ch * 8) * 2));
    }
  };

  // Q A-fragments for d-chunks [0,32) and [32,64)
  v16h qf0, qf1;
  {
    const _Float16* p = qbase + (qt + l15) * 64 + lhalf * 8;
#pragma unroll
    for (int i = 0; i < 8; ++i) {
      qf0[i] = p[i];       qf0[8 + i] = p[16 + i];
      qf1[i] = p[32 + i];  qf1[8 + i] = p[48 + i];
    }
  }

  float mrow[8], lrow[8];
  v8f acc[4];
#pragma unroll
  for (int j = 0; j < 8; ++j) { mrow[j] = -INFINITY; lrow[j] = 0.0f; }
#pragma unroll
  for (int t = 0; t < 4; ++t)
#pragma unroll
    for (int j = 0; j < 8; ++j) acc[t][j] = 0.0f;

  const int kend = qb * 64 + 64;
  stage(0, 0);
  wait_async_all();
  __syncthreads();

  for (int kb = 0; kb < kend; kb += 32) {
    const int cur = (kb >> 5) & 1;
    if (kb + 32 < kend) stage(kb + 32, cur ^ 1);

    if (kb <= qt + 15) {                   // wave-uniform: EXEC stays all-ones
      v8f s[2];
#pragma unroll
      for (int g = 0; g < 2; ++g) {
#pragma unroll
        for (int j = 0; j < 8; ++j) s[g][j] = 0.0f;
#pragma unroll
        for (int dc = 0; dc < 2; ++dc) {
          v16h bf;
          const _Float16* p = &Klds[cur][(g * 16 + l15) * 72 + dc * 32 + lhalf * 8];
#pragma unroll
          for (int i = 0; i < 8; ++i) { bf[i] = p[i]; bf[8 + i] = p[16 + i]; }
          s[g] = WMMA_F32_F16(dc == 0 ? qf0 : qf1, bf, s[g]);
        }
      }

      // causal mask + online softmax (C layout: lane=col l15, rows lhalf*8+j)
      float curm[8];
#pragma unroll
      for (int j = 0; j < 8; ++j) {
        int tq = qt + lhalf * 8 + j;
#pragma unroll
        for (int g = 0; g < 2; ++g) {
          int tk = kb + g * 16 + l15;
          if (tk > tq) s[g][j] = -INFINITY;
        }
        curm[j] = fmaxf(s[0][j], s[1][j]);
      }
#pragma unroll
      for (int msk = 1; msk < 16; msk <<= 1)
#pragma unroll
        for (int j = 0; j < 8; ++j)
          curm[j] = fmaxf(curm[j], __shfl_xor(curm[j], msk, 32));

      float alpha[8], rsum[8];
#pragma unroll
      for (int j = 0; j < 8; ++j) {
        float nm = fmaxf(mrow[j], curm[j]);
        alpha[j] = __expf(mrow[j] - nm);
        mrow[j] = nm;
        float p0 = __expf(s[0][j] - nm);
        float p1 = __expf(s[1][j] - nm);
        s[0][j] = p0; s[1][j] = p1;
        rsum[j] = p0 + p1;
      }
#pragma unroll
      for (int msk = 1; msk < 16; msk <<= 1)
#pragma unroll
        for (int j = 0; j < 8; ++j)
          rsum[j] += __shfl_xor(rsum[j], msk, 32);
#pragma unroll
      for (int j = 0; j < 8; ++j) lrow[j] = lrow[j] * alpha[j] + rsum[j];
#pragma unroll
      for (int t = 0; t < 4; ++t)
#pragma unroll
        for (int j = 0; j < 8; ++j) acc[t][j] *= alpha[j];

      // reshape P: C layout -> A layout via per-wave LDS
      _Float16* pl = &Plds[wave][0];
#pragma unroll
      for (int g = 0; g < 2; ++g)
#pragma unroll
        for (int j = 0; j < 8; ++j)
          pl[(lhalf * 8 + j) * 48 + g * 16 + l15] = (_Float16)s[g][j];
      asm volatile("s_wait_dscnt 0" ::: "memory");  // in-wave LDS store->load

      v16h pf;
      {
        const _Float16* p = &pl[l15 * 48 + lhalf * 8];
#pragma unroll
        for (int i = 0; i < 8; ++i) { pf[i] = p[i]; pf[8 + i] = p[16 + i]; }
      }
#pragma unroll
      for (int t = 0; t < 4; ++t) {       // O += P @ V
        v16h vf;
        const _Float16* p = &Vlds[cur][(t * 16 + l15) * 48 + lhalf * 8];
#pragma unroll
        for (int i = 0; i < 8; ++i) { vf[i] = p[i]; vf[8 + i] = p[16 + i]; }
        acc[t] = WMMA_F32_F16(pf, vf, acc[t]);
      }
    }
    wait_async_all();
    __syncthreads();
  }

  // normalize and store to [B,T,C] f16
  const int bb = bh >> 4, h = bh & 15;
#pragma unroll
  for (int j = 0; j < 8; ++j) {
    int tq = qt + lhalf * 8 + j;
    float inv = 1.0f / lrow[j];
#pragma unroll
    for (int t = 0; t < 4; ++t) {
      int d = t * 16 + l15;
      attn_out[((size_t)bb * 2048 + tq) * 1024 + h * 64 + d] = (_Float16)(acc[t][j] * inv);
    }
  }
}

// ---------------------------------------------------------------------------
// Kernel 3: out = attn @ w_proj + b_proj (fp32 out). Same pipeline as k1.
// ---------------------------------------------------------------------------
__global__ __launch_bounds__(256) void proj_gemm_kernel(
    const _Float16* __restrict__ a,    // [8192,1024] f16
    const _Float16* __restrict__ wT,   // [1024,1024] f16 (n-major)
    const float* __restrict__ bias,    // [1024]
    float* __restrict__ out)           // [8192,1024] fp32
{
  __shared__ __align__(16) _Float16 Alds[2][64 * 48];
  __shared__ __align__(16) _Float16 Blds[2][128 * 48];

  const int tid   = threadIdx.x;
  const int lane  = tid & 31;
  const int wave  = tid >> 5;
  const int m0    = blockIdx.y * 64;
  const int n0    = blockIdx.x * 128;
  const int mw    = (wave >> 1) * 16;
  const int nw    = (wave & 1) * 64;
  const int lhalf = lane >> 4;
  const int l15   = lane & 15;

  const unsigned long long as = (unsigned long long)(size_t)a;
  const unsigned long long wb = (unsigned long long)(size_t)wT;
  const unsigned aoff0 = lds_off_of(&Alds[0][0]);
  const unsigned boff0 = lds_off_of(&Blds[0][0]);

  auto stage = [&](int k0, int b) {
    {
      int r = tid >> 2, cg = tid & 3;
      async_copy_b128(aoff0 + (unsigned)(b * 6144 + r * 96 + cg * 16), as,
                      (unsigned)(((m0 + r) * 1024 + k0 + cg * 8) * 2));
    }
#pragma unroll
    for (int it = 0; it < 2; ++it) {
      int s = tid + it * 256;
      int n = s >> 2, cg = s & 3;
      async_copy_b128(boff0 + (unsigned)(b * 12288 + n * 96 + cg * 16), wb,
                      (unsigned)(((n0 + n) * 1024 + k0 + cg * 8) * 2));
    }
  };

  v8f acc[4];
#pragma unroll
  for (int t = 0; t < 4; ++t)
#pragma unroll
    for (int j = 0; j < 8; ++j) acc[t][j] = 0.0f;

  stage(0, 0);
  wait_async_all();
  __syncthreads();

  for (int kt = 0; kt < 32; ++kt) {
    const int cur = kt & 1;
    if (kt + 1 < 32) stage((kt + 1) * 32, cur ^ 1);

    const _Float16* Ab = &Alds[cur][0];
    const _Float16* Bb = &Blds[cur][0];
    v16h af;
    {
      const _Float16* p = &Ab[(mw + l15) * 48 + lhalf * 8];
#pragma unroll
      for (int i = 0; i < 8; ++i) { af[i] = p[i]; af[8 + i] = p[16 + i]; }
    }
#pragma unroll
    for (int t = 0; t < 4; ++t) {
      v16h bf;
      const _Float16* p = &Bb[(nw + t * 16 + l15) * 48 + lhalf * 8];
#pragma unroll
      for (int i = 0; i < 8; ++i) { bf[i] = p[i]; bf[8 + i] = p[16 + i]; }
      acc[t] = WMMA_F32_F16(af, bf, acc[t]);
    }
    wait_async_all();
    __syncthreads();
  }

#pragma unroll
  for (int t = 0; t < 4; ++t) {
#pragma unroll
    for (int j = 0; j < 8; ++j) {
      int m = m0 + mw + lhalf * 8 + j;
      int n = n0 + nw + t * 16 + l15;
      out[(size_t)m * 1024 + n] = acc[t][j] + bias[n];
    }
  }
}

// ---------------------------------------------------------------------------
extern "C" void kernel_launch(void* const* d_in, const int* in_sizes, int n_in,
                              void* d_out, int out_size, void* d_ws, size_t ws_size,
                              hipStream_t stream) {
  (void)in_sizes; (void)n_in; (void)out_size; (void)ws_size;
  const float* x      = (const float*)d_in[0];
  const float* w_attn = (const float*)d_in[1];
  const float* b_attn = (const float*)d_in[2];
  const float* w_proj = (const float*)d_in[3];
  const float* b_proj = (const float*)d_in[4];
  float* out = (float*)d_out;

  char* ws = (char*)d_ws;
  _Float16* qh    = (_Float16*)(ws);                     // 16 MB  q [B,H,T,D]
  _Float16* kh    = (_Float16*)(ws + (16ull << 20));     // 16 MB  k [B,H,T,D]
  _Float16* vh    = (_Float16*)(ws + (32ull << 20));     // 16 MB  v [B,H,D,T]
  _Float16* x16ah = (_Float16*)(ws + (48ull << 20));     // 16 MB  x16, reused as attn out
  _Float16* wa16T = (_Float16*)(ws + (64ull << 20));     //  6 MB  w_attn^T f16
  _Float16* wp16T = (_Float16*)(ws + (70ull << 20));     //  2 MB  w_proj^T f16

  cvt_f16_kernel<<<8192, 256, 0, stream>>>(x, x16ah);                          // 8.39M elems
  cvt_transpose_kernel<<<dim3(96, 32), 256, 0, stream>>>(w_attn, wa16T, 1024, 3072);
  cvt_transpose_kernel<<<dim3(32, 32), 256, 0, stream>>>(w_proj, wp16T, 1024, 1024);

  qkv_gemm_kernel<<<dim3(24, 128), 256, 0, stream>>>(x16ah, wa16T, b_attn, qh, kh, vh);
  flash_attn_kernel<<<dim3(32, 64), 128, 0, stream>>>(qh, kh, vh, x16ah);
  proj_gemm_kernel<<<dim3(8, 128), 256, 0, stream>>>(x16ah, wp16T, b_proj, out);
}
